// GAttNHP_Model_87179246174577
// MI455X (gfx1250) — matrix operand order repaired and checked by reading, not compiled, not run on hardware
//
#include <hip/hip_runtime.h>

typedef _Float16 half_t;
typedef __attribute__((ext_vector_type(16))) _Float16 v16h;
typedef __attribute__((ext_vector_type(8)))  float    v8f;

// ---------------------------------------------------------------------------
// WMMA helpers (CDNA5 wave32, V_WMMA_F32_16X16X32_F16)
// A-frag (16x32 f16): lane l: m=l&15, kb=(l>>4)*16 -> halfs [kb..kb+15] (contig 32B)
// B-frag (32x16 f16): lane l: n=l&15, reads Bt[n][k0+(l>>4)*16 .. +15] (contig 32B)
// C/D  (16x16 f32):   lane l vgpr r: row = r+8*(l>>4), col = l&15
// ---------------------------------------------------------------------------
__device__ __forceinline__ v8f wmma_f16(v16h a, v16h b, v8f c) {
  return __builtin_amdgcn_wmma_f32_16x16x32_f16(false, a, false, b, (short)0, c,
                                                false, false);
}
__device__ __forceinline__ v16h load_afrag(const half_t* p) {
  v16h a;
  ((uint4*)&a)[0] = *(const uint4*)(p);
  ((uint4*)&a)[1] = *(const uint4*)(p + 8);
  return a;
}
__device__ __forceinline__ v16h load_bfrag(const half_t* p) {
  v16h b;
  ((uint4*)&b)[0] = *(const uint4*)(p);
  ((uint4*)&b)[1] = *(const uint4*)(p + 8);
  return b;
}

// ---------------------------------------------------------------------------
// Software-pipelined 32x64 tile core: acc[2][4], 8 WMMAs per K-chunk, next
// chunk's fragments loaded before current chunk's WMMAs issue.
// ---------------------------------------------------------------------------
__device__ __forceinline__ void gemm_core_32x64(
    const half_t* __restrict__ A, const half_t* __restrict__ Bt,
    int rowbase, int colbase, int K, int lda, int ldb, int lane, v8f acc[2][4]) {
  int m = lane & 15, lh = lane >> 4;
  const half_t* ap0 = A + (size_t)(rowbase + m) * lda + lh * 16;
  const half_t* ap1 = ap0 + (size_t)16 * lda;
  const half_t* bp  = Bt + (size_t)(colbase + m) * ldb + lh * 16;
  size_t bs = (size_t)16 * ldb;

  v16h a0 = load_afrag(ap0), a1 = load_afrag(ap1);
  v16h b0 = load_bfrag(bp), b1 = load_bfrag(bp + bs);
  v16h b2 = load_bfrag(bp + 2 * bs), b3 = load_bfrag(bp + 3 * bs);

  for (int k0 = 32; k0 < K; k0 += 32) {
    v16h na0 = load_afrag(ap0 + k0), na1 = load_afrag(ap1 + k0);
    v16h nb0 = load_bfrag(bp + k0), nb1 = load_bfrag(bp + bs + k0);
    v16h nb2 = load_bfrag(bp + 2 * bs + k0), nb3 = load_bfrag(bp + 3 * bs + k0);
    acc[0][0] = wmma_f16(a0, b0, acc[0][0]);
    acc[1][0] = wmma_f16(a1, b0, acc[1][0]);
    acc[0][1] = wmma_f16(a0, b1, acc[0][1]);
    acc[1][1] = wmma_f16(a1, b1, acc[1][1]);
    acc[0][2] = wmma_f16(a0, b2, acc[0][2]);
    acc[1][2] = wmma_f16(a1, b2, acc[1][2]);
    acc[0][3] = wmma_f16(a0, b3, acc[0][3]);
    acc[1][3] = wmma_f16(a1, b3, acc[1][3]);
    a0 = na0; a1 = na1; b0 = nb0; b1 = nb1; b2 = nb2; b3 = nb3;
  }
  acc[0][0] = wmma_f16(a0, b0, acc[0][0]);
  acc[1][0] = wmma_f16(a1, b0, acc[1][0]);
  acc[0][1] = wmma_f16(a0, b1, acc[0][1]);
  acc[1][1] = wmma_f16(a1, b1, acc[1][1]);
  acc[0][2] = wmma_f16(a0, b2, acc[0][2]);
  acc[1][2] = wmma_f16(a1, b2, acc[1][2]);
  acc[0][3] = wmma_f16(a0, b3, acc[0][3]);
  acc[1][3] = wmma_f16(a1, b3, acc[1][3]);
}

// ---------------------------------------------------------------------------
// Transpose-convert f32 [K][N] -> f16 Bt [N][K]
// ---------------------------------------------------------------------------
__global__ void __launch_bounds__(256) tconv_kernel(const float* __restrict__ src,
                                                    half_t* __restrict__ dst,
                                                    int K, int N) {
  int idx = blockIdx.x * 256 + threadIdx.x;
  if (idx >= K * N) return;
  int k = idx / N, n = idx % N;
  dst[(size_t)n * K + k] = (half_t)src[idx];
}

// ---------------------------------------------------------------------------
// Build x (obj_embed + time encodings), cur0 (time encoding), s/r feat columns.
// Token rows padded: row = b*512 + i, i==511 is a zero pad row.
// ---------------------------------------------------------------------------
__global__ void __launch_bounds__(256) build_kernel(
    const int* __restrict__ subs, const int* __restrict__ marks,
    const int* __restrict__ objs, const float* __restrict__ times,
    const float* __restrict__ dts, const float* __restrict__ obj_embed,
    const float* __restrict__ sub_embed, const float* __restrict__ rel_embed,
    half_t* __restrict__ xh, float* __restrict__ curf, half_t* __restrict__ curh,
    float* __restrict__ feats) {
  int row = blockIdx.x;
  int d = threadIdx.x;          // 0..255
  int b = row >> 9, i = row & 511;
  int j = d & 127;
  float freq = __expf(-0.07195578415606393f * (float)j);  // 10000^(-j/128)
  float xv = 0.f, cv = 0.f;
  if (i < 511) {
    float tx  = times[b * 512 + i];
    float dtx = dts[b * 512 + i];
    float tc  = times[b * 512 + i + 1];
    int o = objs[b * 512 + i];
    float ex = (d < 128) ? sinf(tx * freq)  : cosf(tx * freq);
    float ed = (d < 128) ? sinf(dtx * freq) : cosf(dtx * freq);
    float ec = (d < 128) ? sinf(tc * freq)  : cosf(tc * freq);
    xv = obj_embed[(size_t)o * 256 + d] + ex + ed;
    cv = ec;
  }
  xh[(size_t)row * 256 + d]   = (half_t)xv;
  curf[(size_t)row * 256 + d] = cv;
  curh[(size_t)row * 256 + d] = (half_t)cv;
  int s0 = subs[b * 512], m0 = marks[b * 512];
  feats[(size_t)row * 1024 + 512 + d] = sub_embed[(size_t)s0 * 256 + d];
  feats[(size_t)row * 1024 + 768 + d] = rel_embed[(size_t)m0 * 256 + d];
}

// ---------------------------------------------------------------------------
// Generic WMMA GEMM: C(f16) = A x Bt + bias. Wave = 32x64 tile.
// Block = 8 waves (2 M x 4 N) -> 64 x 256 block tile.
// ---------------------------------------------------------------------------
__global__ void __launch_bounds__(256) gemm_h_kernel(
    const half_t* __restrict__ A, const half_t* __restrict__ Bt,
    half_t* __restrict__ C, const float* __restrict__ bias,
    int M, int N, int K, int lda, int ldb, int ldc) {
  int wid = threadIdx.x >> 5, lane = threadIdx.x & 31;
  int rowbase = blockIdx.x * 64 + (wid >> 2) * 32;
  int colbase = blockIdx.y * 256 + (wid & 3) * 64;
  if (rowbase >= M || colbase >= N) return;
  int m = lane & 15, lh = lane >> 4;
  v8f acc[2][4] = {};
  gemm_core_32x64(A, Bt, rowbase, colbase, K, lda, ldb, lane, acc);
#pragma unroll
  for (int rt = 0; rt < 2; rt++) {
#pragma unroll
    for (int t = 0; t < 4; t++) {
#pragma unroll
      for (int r = 0; r < 8; r++) {
        int row = rowbase + rt * 16 + r + 8 * lh;
        int col = colbase + 16 * t + m;
        float v = acc[rt][t][r] + (bias ? bias[col] : 0.f);
        C[(size_t)row * ldc + col] = (half_t)v;
      }
    }
  }
}

// ---------------------------------------------------------------------------
// O-projection GEMM with residual epilogue:
//   cur += attn @ WoT ; feats[:, lofs:lofs+256] = cur ; curh = f16(cur)
// M=8192, N=256, K=256 fixed strides.
// ---------------------------------------------------------------------------
__global__ void __launch_bounds__(256) gemm_oproj_kernel(
    const half_t* __restrict__ A, const half_t* __restrict__ Bt,
    float* __restrict__ curf, half_t* __restrict__ curh,
    float* __restrict__ feats, int lofs) {
  int wid = threadIdx.x >> 5, lane = threadIdx.x & 31;
  int rowbase = blockIdx.x * 64 + (wid >> 2) * 32;
  int colbase = (wid & 3) * 64;
  int m = lane & 15, lh = lane >> 4;
  v8f acc[2][4] = {};
  gemm_core_32x64(A, Bt, rowbase, colbase, 256, 256, 256, lane, acc);
#pragma unroll
  for (int rt = 0; rt < 2; rt++) {
#pragma unroll
    for (int t = 0; t < 4; t++) {
#pragma unroll
      for (int r = 0; r < 8; r++) {
        int row = rowbase + rt * 16 + r + 8 * lh;
        int col = colbase + 16 * t + m;
        float v = acc[rt][t][r] + curf[(size_t)row * 256 + col];
        curf[(size_t)row * 256 + col] = v;
        curh[(size_t)row * 256 + col] = (half_t)v;
        feats[(size_t)row * 1024 + lofs + col] = v;
      }
    }
  }
}

// ---------------------------------------------------------------------------
// Final GEMM: out = softplus(enh @ intWt + int_b), store only rows i<511.
// ---------------------------------------------------------------------------
__global__ void __launch_bounds__(256) gemm_final_kernel(
    const half_t* __restrict__ A, const half_t* __restrict__ Bt,
    const float* __restrict__ bias, float* __restrict__ out,
    int M, int N, int K) {
  int wid = threadIdx.x >> 5, lane = threadIdx.x & 31;
  int rowbase = blockIdx.x * 64 + (wid >> 2) * 32;
  int colbase = blockIdx.y * 256 + (wid & 3) * 64;
  if (rowbase >= M || colbase >= N) return;
  int m = lane & 15, lh = lane >> 4;
  v8f acc[2][4] = {};
  gemm_core_32x64(A, Bt, rowbase, colbase, 1024, 1024, 1024, lane, acc);
#pragma unroll
  for (int rt = 0; rt < 2; rt++) {
#pragma unroll
    for (int t = 0; t < 4; t++) {
#pragma unroll
      for (int r = 0; r < 8; r++) {
        int row = rowbase + rt * 16 + r + 8 * lh;
        int i = row & 511, b = row >> 9;
        if (i >= 511) continue;
        int col = colbase + 16 * t + m;
        float v = acc[rt][t][r] + bias[col];
        float sp = (v > 20.f) ? v : log1pf(__expf(v));
        out[((size_t)(b * 511 + i)) * 8000 + col] = sp;
      }
    }
  }
}

// ---------------------------------------------------------------------------
// vT[b][h][d][j] = vh[b*512+j][h*64+d]  (so PV B-fragments are contiguous)
// ---------------------------------------------------------------------------
__global__ void __launch_bounds__(256) vtrans_kernel(const half_t* __restrict__ vh,
                                                     half_t* __restrict__ vT) {
  int idx = blockIdx.x * 256 + threadIdx.x;  // 8192*256
  int row = idx >> 8, c = idx & 255;
  int b = row >> 9, i = row & 511, h = c >> 6, d = c & 63;
  vT[(((size_t)(b * 4 + h) * 64) + d) * 512 + i] = vh[idx];
}

// ---------------------------------------------------------------------------
// Causal flash attention, one wave per (b, h, 16-query-row tile).
// d=64, key tiles of 32, online softmax; QK^T and PV via WMMA.
// V fragments are prefetched before the softmax VALU work.
// ---------------------------------------------------------------------------
__global__ void __launch_bounds__(256) attn_kernel(
    const half_t* __restrict__ qh, const half_t* __restrict__ kh,
    const half_t* __restrict__ vT, half_t* __restrict__ attnh) {
  __shared__ half_t Pt[8][16 * 32];  // per-wave P staging tile
  int wid = threadIdx.x >> 5, lane = threadIdx.x & 31;
  int w = blockIdx.x * 8 + wid;      // 2048 waves = 16b * 4h * 32qt
  int qt = w & 31, h = (w >> 5) & 3, b = w >> 7;
  int m = lane & 15, lh = lane >> 4;
  int qrow = b * 512 + qt * 16;

  const half_t* qp = qh + (size_t)(qrow + m) * 256 + h * 64 + lh * 16;
  v16h a0 = load_afrag(qp);        // d 0..31
  v16h a1 = load_afrag(qp + 32);   // d 32..63

  v8f o[4] = {};
  float mst[8], lst[8];
#pragma unroll
  for (int r = 0; r < 8; r++) { mst[r] = -1e30f; lst[r] = 0.f; }

  const half_t* kbase = kh + (size_t)(b * 512) * 256 + h * 64;
  const half_t* vbase = vT + ((size_t)(b * 4 + h) * 64) * 512;
  int ktmax = (qt * 16 + 15) >> 5;
  for (int kt = 0; kt <= ktmax; kt++) {
    int jb = kt * 32;
    // S = Q K^T for keys [jb, jb+32): load all 4 K-frags, then 4 WMMAs
    const half_t* kp  = kbase + (size_t)(jb + m) * 256 + lh * 16;
    const half_t* kp2 = kbase + (size_t)(jb + 16 + m) * 256 + lh * 16;
    v16h bk00 = load_bfrag(kp),  bk01 = load_bfrag(kp + 32);
    v16h bk10 = load_bfrag(kp2), bk11 = load_bfrag(kp2 + 32);
    v8f s0 = {}, s1 = {};
    s0 = wmma_f16(a0, bk00, s0);
    s0 = wmma_f16(a1, bk01, s0);
    s1 = wmma_f16(a0, bk10, s1);
    s1 = wmma_f16(a1, bk11, s1);
    // Prefetch V B-frags; they complete under the softmax VALU work below.
    const half_t* vb = vbase + jb + lh * 16;
    v16h vb0 = load_bfrag(vb + (size_t)(m) * 512);
    v16h vb1 = load_bfrag(vb + (size_t)(16 + m) * 512);
    v16h vb2 = load_bfrag(vb + (size_t)(32 + m) * 512);
    v16h vb3 = load_bfrag(vb + (size_t)(48 + m) * 512);
    // online softmax update (per C-layout row), write P to LDS in A-layout order
#pragma unroll
    for (int r = 0; r < 8; r++) {
      int qi = qt * 16 + r + 8 * lh;
      float v0 = s0[r] * 0.125f; if (jb + m > qi) v0 = -1e9f;
      float v1 = s1[r] * 0.125f; if (jb + 16 + m > qi) v1 = -1e9f;
      float pm = fmaxf(v0, v1);
      for (int off = 1; off < 16; off <<= 1) pm = fmaxf(pm, __shfl_xor(pm, off, 32));
      float nm = fmaxf(mst[r], pm);
      float corr = __expf(mst[r] - nm);
      float p0 = __expf(v0 - nm), p1 = __expf(v1 - nm);
      float rs = p0 + p1;
      for (int off = 1; off < 16; off <<= 1) rs += __shfl_xor(rs, off, 32);
      lst[r] = lst[r] * corr + rs;
      mst[r] = nm;
      o[0][r] *= corr; o[1][r] *= corr; o[2][r] *= corr; o[3][r] *= corr;
      Pt[wid][(r + 8 * lh) * 32 + m]      = (half_t)p0;
      Pt[wid][(r + 8 * lh) * 32 + 16 + m] = (half_t)p1;
    }
    // O += P @ V  (P 16x32 from LDS as A-frag)
    v16h pa = load_afrag(&Pt[wid][m * 32 + lh * 16]);
    o[0] = wmma_f16(pa, vb0, o[0]);
    o[1] = wmma_f16(pa, vb1, o[1]);
    o[2] = wmma_f16(pa, vb2, o[2]);
    o[3] = wmma_f16(pa, vb3, o[3]);
  }
#pragma unroll
  for (int t = 0; t < 4; t++) {
#pragma unroll
    for (int r = 0; r < 8; r++) {
      int row = qrow + r + 8 * lh;
      int col = h * 64 + 16 * t + m;
      attnh[(size_t)row * 256 + col] = (half_t)(o[t][r] / lst[r]);
    }
  }
}

// ---------------------------------------------------------------------------
// Masked segment mean: rep[b][g][f] over tokens. Block = (b, 128-feature chunk).
// ---------------------------------------------------------------------------
__global__ void __launch_bounds__(128) seg_kernel(
    const float* __restrict__ feats, const int* __restrict__ subs,
    const int* __restrict__ marks, const unsigned char* __restrict__ mask,
    const int* __restrict__ group_map, float* __restrict__ rep) {
  __shared__ float acc[64 * 128];
  __shared__ float cnt[64];
  int b = blockIdx.x, fc = blockIdx.y, tid = threadIdx.x;
  for (int idx = tid; idx < 64 * 128; idx += 128) acc[idx] = 0.f;
  if (tid < 64) cnt[tid] = 0.f;
  __syncthreads();
  for (int i = 0; i < 511; i++) {
    if (mask[b * 512 + i]) {
      int g = group_map[subs[b * 512 + i] * 100 + marks[b * 512 + i]];
      acc[g * 128 + tid] += feats[((size_t)(b * 512 + i)) * 1024 + fc * 128 + tid];
      if (tid == 0) cnt[g] += 1.f;
    }
  }
  __syncthreads();
  for (int idx = tid; idx < 64 * 128; idx += 128) {
    int g = idx >> 7, f = idx & 127;
    rep[((size_t)(b * 64 + g)) * 1024 + fc * 128 + f] = acc[idx] / fmaxf(cnt[g], 1.f);
  }
}

// ---------------------------------------------------------------------------
// Small f32 matmul: Y[r][n] = bias[n] + sum_k X[r][k] W[k][n] (+res, relu)
// ---------------------------------------------------------------------------
__global__ void __launch_bounds__(256) smm_kernel(
    const float* __restrict__ X, const float* __restrict__ W,
    const float* __restrict__ bias, const float* __restrict__ res,
    float* __restrict__ Y, int R, int Kd, int Nn, int relu) {
  int idx = blockIdx.x * 256 + threadIdx.x;
  if (idx >= R * Nn) return;
  int r = idx / Nn, n = idx % Nn;
  float acc = bias ? bias[n] : 0.f;
  const float* x = X + (size_t)r * Kd;
  for (int k = 0; k < Kd; k++) acc += x[k] * W[(size_t)k * Nn + n];
  if (res) acc += res[idx];
  if (relu) acc = fmaxf(acc, 0.f);
  Y[idx] = acc;
}

// ---------------------------------------------------------------------------
// Group attention (2 heads, hd=32, no mask). Block = (b, h).
// ---------------------------------------------------------------------------
__global__ void __launch_bounds__(256) gattn_kernel(const float* __restrict__ qkv,
                                                    float* __restrict__ gapre) {
  __shared__ float sq[64 * 32], sk[64 * 32], sv[64 * 32], sa[64 * 64];
  int b = blockIdx.x, h = blockIdx.y, tid = threadIdx.x;
  for (int idx = tid; idx < 2048; idx += 256) {
    int g = idx >> 5, d = idx & 31;
    size_t base = ((size_t)(b * 64 + g)) * 192 + h * 32 + d;
    sq[idx] = qkv[base];
    sk[idx] = qkv[base + 64];
    sv[idx] = qkv[base + 128];
  }
  __syncthreads();
  for (int idx = tid; idx < 4096; idx += 256) {
    int g1 = idx >> 6, g2 = idx & 63;
    float a = 0.f;
    for (int d = 0; d < 32; d++) a += sq[g1 * 32 + d] * sk[g2 * 32 + d];
    sa[idx] = a * 0.17677669529663687f;  // 1/sqrt(32)
  }
  __syncthreads();
  if (tid < 64) {
    float mx = -1e30f;
    for (int j2 = 0; j2 < 64; j2++) mx = fmaxf(mx, sa[tid * 64 + j2]);
    float sm = 0.f;
    for (int j2 = 0; j2 < 64; j2++) { float e = __expf(sa[tid * 64 + j2] - mx); sa[tid * 64 + j2] = e; sm += e; }
    float inv = 1.f / sm;
    for (int j2 = 0; j2 < 64; j2++) sa[tid * 64 + j2] *= inv;
  }
  __syncthreads();
  for (int idx = tid; idx < 2048; idx += 256) {
    int g1 = idx >> 5, d = idx & 31;
    float a = 0.f;
    for (int g2 = 0; g2 < 64; g2++) a += sa[g1 * 64 + g2] * sv[g2 * 32 + d];
    gapre[((size_t)(b * 64 + g1)) * 64 + h * 32 + d] = a;
  }
}

// ---------------------------------------------------------------------------
// Row LayerNorm over 64 features (one thread per row).
// ---------------------------------------------------------------------------
__global__ void __launch_bounds__(256) ln_kernel(const float* __restrict__ X,
                                                 const float* __restrict__ w,
                                                 const float* __restrict__ bb,
                                                 float* __restrict__ Y, int rows) {
  int r = blockIdx.x * 256 + threadIdx.x;
  if (r >= rows) return;
  const float* x = X + (size_t)r * 64;
  float mu = 0.f;
  for (int p = 0; p < 64; p++) mu += x[p];
  mu *= (1.f / 64.f);
  float var = 0.f;
  for (int p = 0; p < 64; p++) { float d = x[p] - mu; var += d * d; }
  var *= (1.f / 64.f);
  float inv = rsqrtf(var + 1e-5f);
  float* y = Y + (size_t)r * 64;
  for (int p = 0; p < 64; p++) y[p] = (x[p] - mu) * inv * w[p] + bb[p];
}

// ---------------------------------------------------------------------------
// merged f16 [8192][1088] = [f16(feats) | gout[b][g]*fm]
// ---------------------------------------------------------------------------
__global__ void __launch_bounds__(256) merged_kernel(
    const float* __restrict__ feats, const float* __restrict__ gout,
    const int* __restrict__ subs, const int* __restrict__ marks,
    const unsigned char* __restrict__ mask, const int* __restrict__ group_map,
    half_t* __restrict__ merged) {
  int row = blockIdx.x, tid = threadIdx.x;
  size_t fo = (size_t)row * 1024, mo = (size_t)row * 1088;
  for (int c = tid; c < 1024; c += 256) merged[mo + c] = (half_t)feats[fo + c];
  if (tid < 64) {
    int b = row >> 9, i = row & 511;
    float v = 0.f;
    if (i < 511 && mask[b * 512 + i]) {
      int g = group_map[subs[b * 512 + i] * 100 + marks[b * 512 + i]];
      v = gout[((size_t)(b * 64 + g)) * 64 + tid];
    }
    merged[mo + 1024 + tid] = (half_t)v;
  }
}

// ---------------------------------------------------------------------------
extern "C" void kernel_launch(void* const* d_in, const int* in_sizes, int n_in,
                              void* d_out, int out_size, void* d_ws, size_t ws_size,
                              hipStream_t stream) {
  (void)in_sizes; (void)n_in; (void)out_size; (void)ws_size;
  const int*   subs      = (const int*)d_in[0];
  const int*   marks     = (const int*)d_in[1];
  const int*   objs      = (const int*)d_in[2];
  const float* times     = (const float*)d_in[3];
  const float* dts       = (const float*)d_in[4];
  const unsigned char* maskp = (const unsigned char*)d_in[5];  // bool array
  const int*   group_map = (const int*)d_in[6];
  const float* obj_embed = (const float*)d_in[7];
  const float* core_Wq   = (const float*)d_in[8];
  const float* core_Wk   = (const float*)d_in[9];
  const float* core_Wv   = (const float*)d_in[10];
  const float* core_Wo   = (const float*)d_in[11];
  const float* sub_embed = (const float*)d_in[12];
  const float* rel_embed = (const float*)d_in[13];
  const float* gp_w      = (const float*)d_in[14];
  const float* gp_b      = (const float*)d_in[15];
  const float* ga_in_w   = (const float*)d_in[16];
  const float* ga_in_b   = (const float*)d_in[17];
  const float* ga_out_w  = (const float*)d_in[18];
  const float* ga_out_b  = (const float*)d_in[19];
  const float* ffn_w1    = (const float*)d_in[20];
  const float* ffn_b1    = (const float*)d_in[21];
  const float* ffn_w2    = (const float*)d_in[22];
  const float* ffn_b2    = (const float*)d_in[23];
  const float* n1_w      = (const float*)d_in[24];
  const float* n1_b      = (const float*)d_in[25];
  const float* n2_w      = (const float*)d_in[26];
  const float* n2_b      = (const float*)d_in[27];
  const float* mg_w      = (const float*)d_in[28];
  const float* mg_b      = (const float*)d_in[29];
  const float* int_w     = (const float*)d_in[30];
  const float* int_b     = (const float*)d_in[31];
  float* out = (float*)d_out;

  // ---- workspace layout ----
  size_t off = 0;
  char* base = (char*)d_ws;
  auto alloc = [&](size_t bytes) -> void* {
    void* p = base + off;
    off += (bytes + 255) & ~(size_t)255;
    return p;
  };
  const size_t TOK = 8192;  // 16 batches * 512 padded rows
  half_t* wqt   = (half_t*)alloc(2 * 256 * 256 * sizeof(half_t));
  half_t* wkt   = (half_t*)alloc(2 * 256 * 256 * sizeof(half_t));
  half_t* wvt   = (half_t*)alloc(2 * 256 * 256 * sizeof(half_t));
  half_t* wot   = (half_t*)alloc(2 * 256 * 256 * sizeof(half_t));
  half_t* mgwt  = (half_t*)alloc((size_t)1088 * 1024 * sizeof(half_t));
  half_t* intwt = (half_t*)alloc((size_t)1024 * 8000 * sizeof(half_t));
  half_t* xh    = (half_t*)alloc(TOK * 256 * sizeof(half_t));
  half_t* curh  = (half_t*)alloc(TOK * 256 * sizeof(half_t));
  half_t* qh    = (half_t*)alloc(TOK * 256 * sizeof(half_t));
  half_t* kh    = (half_t*)alloc(TOK * 256 * sizeof(half_t));
  half_t* vh    = (half_t*)alloc(TOK * 256 * sizeof(half_t));
  half_t* vt    = (half_t*)alloc(TOK * 256 * sizeof(half_t));
  half_t* at    = (half_t*)alloc(TOK * 256 * sizeof(half_t));
  half_t* merged= (half_t*)alloc(TOK * 1088 * sizeof(half_t));
  half_t* enh   = (half_t*)alloc(TOK * 1024 * sizeof(half_t));
  float*  curf  = (float*)alloc(TOK * 256 * sizeof(float));
  float*  feats = (float*)alloc(TOK * 1024 * sizeof(float));
  float*  rep   = (float*)alloc((size_t)16 * 64 * 1024 * sizeof(float));
  float*  gpb   = (float*)alloc(65536 * sizeof(float));
  float*  qkvb  = (float*)alloc(196608 * sizeof(float));
  float*  gapre = (float*)alloc(65536 * sizeof(float));
  float*  x1b   = (float*)alloc(65536 * sizeof(float));
  float*  gnb   = (float*)alloc(65536 * sizeof(float));
  float*  ffb   = (float*)alloc(65536 * sizeof(float));
  float*  x2b   = (float*)alloc(65536 * sizeof(float));
  float*  goutb = (float*)alloc(65536 * sizeof(float));

  // ---- weight conversion (f32 [K][N] -> f16 Bt [N][K]) ----
  for (int l = 0; l < 2; l++) {
    tconv_kernel<<<256, 256, 0, stream>>>(core_Wq + l * 65536, wqt + l * 65536, 256, 256);
    tconv_kernel<<<256, 256, 0, stream>>>(core_Wk + l * 65536, wkt + l * 65536, 256, 256);
    tconv_kernel<<<256, 256, 0, stream>>>(core_Wv + l * 65536, wvt + l * 65536, 256, 256);
    tconv_kernel<<<256, 256, 0, stream>>>(core_Wo + l * 65536, wot + l * 65536, 256, 256);
  }
  tconv_kernel<<<(1088 * 1024 + 255) / 256, 256, 0, stream>>>(mg_w, mgwt, 1088, 1024);
  tconv_kernel<<<(1024 * 8000 + 255) / 256, 256, 0, stream>>>(int_w, intwt, 1024, 8000);

  // ---- embeddings + time encodings ----
  build_kernel<<<8192, 256, 0, stream>>>(subs, marks, objs, times, dts, obj_embed,
                                         sub_embed, rel_embed, xh, curf, curh, feats);

  // ---- transformer layers ----
  for (int l = 0; l < 2; l++) {
    gemm_h_kernel<<<dim3(128, 1), 256, 0, stream>>>(curh, wqt + l * 65536, qh, nullptr,
                                                    8192, 256, 256, 256, 256, 256);
    gemm_h_kernel<<<dim3(128, 1), 256, 0, stream>>>(xh, wkt + l * 65536, kh, nullptr,
                                                    8192, 256, 256, 256, 256, 256);
    gemm_h_kernel<<<dim3(128, 1), 256, 0, stream>>>(xh, wvt + l * 65536, vh, nullptr,
                                                    8192, 256, 256, 256, 256, 256);
    vtrans_kernel<<<8192, 256, 0, stream>>>(vh, vt);
    attn_kernel<<<256, 256, 0, stream>>>(qh, kh, vt, at);
    gemm_oproj_kernel<<<dim3(128, 1), 256, 0, stream>>>(at, wot + l * 65536, curf, curh,
                                                        feats, l * 256);
  }

  // ---- masked segment mean ----
  seg_kernel<<<dim3(16, 8), 128, 0, stream>>>(feats, subs, marks, maskp, group_map, rep);

  // ---- group transformer (tiny, f32) ----
  smm_kernel<<<256, 256, 0, stream>>>(rep, gp_w, gp_b, nullptr, gpb, 1024, 1024, 64, 0);
  smm_kernel<<<768, 256, 0, stream>>>(gpb, ga_in_w, ga_in_b, nullptr, qkvb, 1024, 64, 192, 0);
  gattn_kernel<<<dim3(16, 2), 256, 0, stream>>>(qkvb, gapre);
  smm_kernel<<<256, 256, 0, stream>>>(gapre, ga_out_w, ga_out_b, gpb, x1b, 1024, 64, 64, 0);
  ln_kernel<<<4, 256, 0, stream>>>(x1b, n1_w, n1_b, gnb, 1024);
  smm_kernel<<<256, 256, 0, stream>>>(gnb, ffn_w1, ffn_b1, nullptr, ffb, 1024, 64, 64, 1);
  smm_kernel<<<256, 256, 0, stream>>>(ffb, ffn_w2, ffn_b2, gnb, x2b, 1024, 64, 64, 0);
  ln_kernel<<<4, 256, 0, stream>>>(x2b, n2_w, n2_b, goutb, 1024);

  // ---- merge + big GEMMs ----
  merged_kernel<<<8192, 256, 0, stream>>>(feats, goutb, subs, marks, maskp, group_map, merged);
  gemm_h_kernel<<<dim3(128, 4), 256, 0, stream>>>(merged, mgwt, enh, mg_b,
                                                  8192, 1024, 1088, 1088, 1088, 1024);
  gemm_final_kernel<<<dim3(128, 32), 256, 0, stream>>>(enh, intwt, int_b, out,
                                                       8192, 8000, 1024);
}
